// PointNetSetAbstractionMsg_70970039599962
// MI455X (gfx1250) — matrix operand order repaired and checked by reading, not compiled, not run on hardware
//
#include <hip/hip_runtime.h>
#include <hip/hip_bf16.h>

typedef __attribute__((ext_vector_type(16))) _Float16 v16h;
typedef __attribute__((ext_vector_type(8)))  _Float16 v8h;
typedef __attribute__((ext_vector_type(8)))  float    v8f;

#define BN_ 16
#define NP_ 4096
#define SN_ 1024
#define EPS_ 1e-5f

// ---------------------------------------------------------------------------
// Farthest point sampling: one block per batch. xyz + running distance in LDS.
// Writes new_xyz directly into d_out[0 .. B*S*3).
// ---------------------------------------------------------------------------
__global__ __launch_bounds__(256) void fps_kernel(const float* __restrict__ xyz,
                                                  float* __restrict__ out_newxyz) {
    extern __shared__ float sm[];
    float* px   = sm;
    float* py   = sm + NP_;
    float* pz   = sm + 2 * NP_;
    float* dist = sm + 3 * NP_;
    float* redv = sm + 4 * NP_;
    int*   redi = (int*)(sm + 4 * NP_ + 256);

    const int b = blockIdx.x;
    const int t = threadIdx.x;

    for (int i = t; i < NP_; i += 256) {
        px[i]   = xyz[(b * NP_ + i) * 3 + 0];
        py[i]   = xyz[(b * NP_ + i) * 3 + 1];
        pz[i]   = xyz[(b * NP_ + i) * 3 + 2];
        dist[i] = 1e10f;
    }
    __syncthreads();

    int far = 0;
    for (int it = 0; it < SN_; ++it) {
        if (t == 0) {
            out_newxyz[(b * SN_ + it) * 3 + 0] = px[far];
            out_newxyz[(b * SN_ + it) * 3 + 1] = py[far];
            out_newxyz[(b * SN_ + it) * 3 + 2] = pz[far];
        }
        const float cx = px[far], cy = py[far], cz = pz[far];
        float lv = -1.0f;
        int   li = 0;
        for (int i = t; i < NP_; i += 256) {
            const float dx = px[i] - cx, dy = py[i] - cy, dz = pz[i] - cz;
            const float d  = dx * dx + dy * dy + dz * dz;
            const float nd = fminf(dist[i], d);
            dist[i] = nd;
            if (nd > lv) { lv = nd; li = i; }
        }
        redv[t] = lv;
        redi[t] = li;
        __syncthreads();
        for (int off = 128; off > 0; off >>= 1) {
            if (t < off) {
                const float ov = redv[t + off];
                const int   oi = redi[t + off];
                if (ov > redv[t] || (ov == redv[t] && oi < redi[t])) {
                    redv[t] = ov;
                    redi[t] = oi;
                }
            }
            __syncthreads();
        }
        far = redi[0];
        __syncthreads();
    }
}

// ---------------------------------------------------------------------------
// Ball query: wave per centroid, batch xyz in LDS, iterative k-th-min select
// with "clamp to nearest when outside radius" (matches top_k + where).
// ---------------------------------------------------------------------------
template <int KS>
__global__ __launch_bounds__(256) void ballq_kernel(const float* __restrict__ xyz,
                                                    const float* __restrict__ newxyz,
                                                    int* __restrict__ gidx, float r2) {
    extern __shared__ float sm[];
    float* px = sm;
    float* py = sm + NP_;
    float* pz = sm + 2 * NP_;

    const int b  = blockIdx.x >> 7;          // S/8 = 128 blocks per batch
    const int sb = (blockIdx.x & 127) * 8;
    const int t  = threadIdx.x;

    for (int i = t; i < NP_; i += 256) {
        px[i] = xyz[(b * NP_ + i) * 3 + 0];
        py[i] = xyz[(b * NP_ + i) * 3 + 1];
        pz[i] = xyz[(b * NP_ + i) * 3 + 2];
    }
    __syncthreads();

    const int wave = t >> 5;
    const int lane = t & 31;
    const int s    = sb + wave;

    const float cx = newxyz[(b * SN_ + s) * 3 + 0];
    const float cy = newxyz[(b * SN_ + s) * 3 + 1];
    const float cz = newxyz[(b * SN_ + s) * 3 + 2];

    unsigned sel[4] = {0u, 0u, 0u, 0u};
    int first = 0;

    for (int kk = 0; kk < KS; ++kk) {
        float mv = 3.4e38f;
        int   mi = 0x7fffffff;
        for (int j = 0; j < 128; ++j) {
            if ((sel[j >> 5] >> (j & 31)) & 1u) continue;
            const int   p  = lane + (j << 5);
            const float dx = px[p] - cx, dy = py[p] - cy, dz = pz[p] - cz;
            const float d  = dx * dx + dy * dy + dz * dz;
            if (d < mv || (d == mv && p < mi)) { mv = d; mi = p; }
        }
        // wave32 all-reduce min with index tie-break
        for (int off = 16; off > 0; off >>= 1) {
            const float ov = __shfl_xor(mv, off, 32);
            const int   oi = __shfl_xor(mi, off, 32);
            if (ov < mv || (ov == mv && oi < mi)) { mv = ov; mi = oi; }
        }
        if ((mi & 31) == lane) {
            const int j = mi >> 5;
            sel[j >> 5] |= 1u << (j & 31);
        }
        if (kk == 0) first = mi;
        const int o = (mv > r2) ? first : mi;
        if (lane == 0) gidx[(b * SN_ + s) * KS + kk] = o;
    }
}

// ---------------------------------------------------------------------------
// Fused layer kernel:  out = feat @ W^T + bias   (pre-BN, fp16, CHANNEL-major:
// act[oc * TOTROWS + row] so each lane's 8 C rows pack into one b128 store).
//   FIRST:  feat gathered from points (float4) / relative xyz, zero-padded
//   else :  feat = relu(BN(prev)) folded into A-tile staging (v8h loads)
// Accumulates per-channel sum / sumsq: register-reduced, 2 LDS atomics/tile.
// ---------------------------------------------------------------------------
template <int KS, int IC, int ICPAD, int OC, bool FIRST>
__global__ __launch_bounds__(256) void gemm_bn_kernel(
    const float* __restrict__ xyz, const float* __restrict__ points,
    const float* __restrict__ newxyz, const int* __restrict__ gidx,
    const _Float16* __restrict__ prevAct, const float* __restrict__ prevStats,
    const float* __restrict__ prevG, const float* __restrict__ prevBe,
    const float* __restrict__ W, const float* __restrict__ bias,
    _Float16* __restrict__ outAct, float* __restrict__ outStats) {

    constexpr int NK = ICPAD / 32;
    constexpr size_t TOT = (size_t)BN_ * SN_ * KS;   // rows in this branch
    __shared__ _Float16 Atile[128][ICPAD + 8];
    __shared__ _Float16 Wt[OC * ICPAD];
    __shared__ float sScale[IC];
    __shared__ float sShift[IC];
    __shared__ float sSum[OC];
    __shared__ float sSq[OC];

    const int t = threadIdx.x;
    const int rowBase = blockIdx.x * 128;

    // BN coefficients of the *previous* layer (folded into input staging)
    if constexpr (!FIRST) {
        const float cnt = (float)(BN_ * SN_ * KS);
        for (int c = t; c < IC; c += 256) {
            const float mu  = prevStats[c] / cnt;
            const float var = prevStats[IC + c] / cnt - mu * mu;
            const float sc  = prevG[c] * rsqrtf(var + EPS_);
            sScale[c] = sc;
            sShift[c] = prevBe[c] - mu * sc;
        }
    }
    // weights -> LDS (transposed access serves B fragments), zero-pad tail
    for (int i = t; i < OC * ICPAD; i += 256) {
        const int oc = i / ICPAD, c = i % ICPAD;
        Wt[i] = (c < IC) ? (_Float16)W[oc * IC + c] : (_Float16)0.0f;
    }
    for (int c = t; c < OC; c += 256) { sSum[c] = 0.0f; sSq[c] = 0.0f; }
    __syncthreads();

    // stage A tile (128 rows x ICPAD)
    if constexpr (FIRST) {
        // 9 chunks per row: 8x float4 point features + 3 relative xyz
        for (int i = t; i < 128 * 9; i += 256) {
            const int r = i / 9, q = i % 9;
            const int row = rowBase + r;
            const int b   = row / (SN_ * KS);
            const int rem = row - b * SN_ * KS;
            const int s   = rem / KS;
            const int idx = gidx[row];
            if (q < 8) {
                const float4 v4 =
                    *(const float4*)&points[((size_t)(b * NP_ + idx)) * 32 + q * 4];
                Atile[r][q * 4 + 0] = (_Float16)v4.x;
                Atile[r][q * 4 + 1] = (_Float16)v4.y;
                Atile[r][q * 4 + 2] = (_Float16)v4.z;
                Atile[r][q * 4 + 3] = (_Float16)v4.w;
            } else {
#pragma unroll
                for (int d = 0; d < 3; ++d) {
                    Atile[r][32 + d] =
                        (_Float16)(xyz[(b * NP_ + idx) * 3 + d] -
                                   newxyz[(b * SN_ + s) * 3 + d]);
                }
            }
        }
        if constexpr (ICPAD > IC) {   // zero the padded channels
            for (int i = t; i < 128 * (ICPAD - IC); i += 256) {
                const int r = i / (ICPAD - IC);
                const int c = IC + i % (ICPAD - IC);
                Atile[r][c] = (_Float16)0.0f;
            }
        }
    } else {
        // channel-major input: 8 consecutive rows per v8h load
        for (int i = t; i < IC * 16; i += 256) {
            const int c  = i / 16;
            const int rb = i % 16;
            __builtin_prefetch(&prevAct[(size_t)c * TOT + rowBase + rb * 8 + 128], 0, 1);
            const v8h x8 = *(const v8h*)&prevAct[(size_t)c * TOT + rowBase + rb * 8];
            const float sc = sScale[c], sh = sShift[c];
#pragma unroll
            for (int j = 0; j < 8; ++j) {
                const float v = fmaxf(sc * (float)x8[j] + sh, 0.0f);
                Atile[rb * 8 + j][c] = (_Float16)v;
            }
        }
    }
    __syncthreads();

    const int wave  = t >> 5;
    const int lane  = t & 31;
    const int ln    = lane & 15;   // row (A) / col (B) within tile
    const int hh    = lane >> 4;   // half-wave
    const int mbase = wave * 16;

    // A fragments for this wave's 16 rows, all K chunks
    v16h af[NK];
#pragma unroll
    for (int kc = 0; kc < NK; ++kc) {
#pragma unroll
        for (int e = 0; e < 16; ++e) {
            const int kk = kc * 32 + (e < 8 ? e : e + 8) + hh * 8;
            af[kc][e] = Atile[mbase + ln][kk];
        }
    }

#pragma unroll
    for (int oct = 0; oct < OC / 16; ++oct) {
        const int oc0 = oct * 16;
        v8f acc = {};
#pragma unroll
        for (int kc = 0; kc < NK; ++kc) {
            v16h bf;
#pragma unroll
            for (int e = 0; e < 16; ++e)
                bf[e] = Wt[(oc0 + ln) * ICPAD + kc * 32 + (e < 8 ? e : e + 8) + hh * 8];
            acc = __builtin_amdgcn_wmma_f32_16x16x32_f16(
                false, af[kc], false, bf, (short)0, acc, false, false);
        }
        const int oc = oc0 + ln;
        const float bv = bias[oc];
        // lane's 8 rows are consecutive: rowBase+mbase+hh*8 .. +7
        float ps = 0.0f, pq = 0.0f;
        v8h pk;
#pragma unroll
        for (int r = 0; r < 8; ++r) {
            const float v = acc[r] + bv;
            pk[r] = (_Float16)v;
            ps += v;
            pq += v * v;
        }
        *(v8h*)&outAct[(size_t)oc * TOT + rowBase + mbase + hh * 8] = pk;
        atomicAdd(&sSum[oc], ps);   // ds_add_f32 (2 per tile per lane)
        atomicAdd(&sSq[oc], pq);
    }
    __syncthreads();
    for (int c = t; c < OC; c += 256) {
        atomicAdd(&outStats[c], sSum[c]);   // global_atomic_add_f32
        atomicAdd(&outStats[OC + c], sSq[c]);
    }
}

// ---------------------------------------------------------------------------
// Final: BN3 + ReLU + max over K samples, write transposed (B, 320, S) slice.
// Channel-major activations: K samples of one channel are contiguous.
// ---------------------------------------------------------------------------
template <int KS, int OC, int CO>
__global__ __launch_bounds__(128) void bn_max_kernel(
    const _Float16* __restrict__ act, const float* __restrict__ stats,
    const float* __restrict__ g, const float* __restrict__ be,
    float* __restrict__ out) {
    constexpr size_t TOT = (size_t)BN_ * SN_ * KS;
    const int bs = blockIdx.x;          // b*S + s
    const int b  = bs / SN_;
    const int s  = bs - b * SN_;
    const int t  = threadIdx.x;
    if (t >= OC) return;

    const float cnt = (float)(BN_ * SN_ * KS);
    const float mu  = stats[t] / cnt;
    const float var = stats[OC + t] / cnt - mu * mu;
    const float sc  = g[t] * rsqrtf(var + EPS_);
    const float sh  = be[t] - mu * sc;

    float m = 0.0f;
    const _Float16* col = &act[(size_t)t * TOT + (size_t)bs * KS];
#pragma unroll
    for (int k8 = 0; k8 < KS / 8; ++k8) {
        const v8h x8 = *(const v8h*)&col[k8 * 8];
#pragma unroll
        for (int j = 0; j < 8; ++j) {
            const float v = fmaxf(sc * (float)x8[j] + sh, 0.0f);
            m = fmaxf(m, v);
        }
    }
    out[BN_ * SN_ * 3 + (b * 320 + CO + t) * SN_ + s] = m;
}

// ---------------------------------------------------------------------------
// Host-side branch driver (template instantiates the full 3-layer pipeline).
// ---------------------------------------------------------------------------
template <int KS, int OC1, int OC2, int OC3, int CO>
static void run_branch(const float* xyz, const float* points, float* dout,
                       const int* gidx, _Float16* actA, _Float16* actB,
                       float* st0, float* st1, float* st2,
                       const float* const* prm, hipStream_t stream) {
    constexpr int blocks = BN_ * SN_ * KS / 128;
    hipMemsetAsync(st0, 0, 2 * OC1 * sizeof(float), stream);
    gemm_bn_kernel<KS, 35, 64, OC1, true><<<blocks, 256, 0, stream>>>(
        xyz, points, dout, gidx, nullptr, nullptr, nullptr, nullptr,
        prm[0], prm[1], actA, st0);
    hipMemsetAsync(st1, 0, 2 * OC2 * sizeof(float), stream);
    gemm_bn_kernel<KS, OC1, OC1, OC2, false><<<blocks, 256, 0, stream>>>(
        nullptr, nullptr, nullptr, nullptr, actA, st0, prm[2], prm[3],
        prm[4], prm[5], actB, st1);
    hipMemsetAsync(st2, 0, 2 * OC3 * sizeof(float), stream);
    gemm_bn_kernel<KS, OC2, OC2, OC3, false><<<blocks, 256, 0, stream>>>(
        nullptr, nullptr, nullptr, nullptr, actB, st1, prm[6], prm[7],
        prm[8], prm[9], actA, st2);
    bn_max_kernel<KS, OC3, CO><<<BN_ * SN_, 128, 0, stream>>>(
        actA, st2, prm[10], prm[11], dout);
}

extern "C" void kernel_launch(void* const* d_in, const int* in_sizes, int n_in,
                              void* d_out, int out_size, void* d_ws, size_t ws_size,
                              hipStream_t stream) {
    const float* xyz    = (const float*)d_in[0];
    const float* points = (const float*)d_in[1];
    float* dout = (float*)d_out;

    // params flattened: 3 branches x 3 layers x (W, b, g, be)
    auto P = [&](int br, int ly, int comp) -> const float* {
        return (const float*)d_in[2 + (br * 3 + ly) * 4 + comp];
    };

    // workspace carve-out
    char* base = (char*)d_ws;
    size_t o = 0;
    auto carve = [&](size_t bytes) -> char* {
        char* p = base + o;
        o += (bytes + 255) & ~(size_t)255;
        return p;
    };
    int* gidx0 = (int*)carve((size_t)BN_ * SN_ * 16 * sizeof(int));
    int* gidx1 = (int*)carve((size_t)BN_ * SN_ * 32 * sizeof(int));
    int* gidx2 = (int*)carve((size_t)BN_ * SN_ * 128 * sizeof(int));
    float* st0 = (float*)carve(1024);
    float* st1 = (float*)carve(1024);
    float* st2 = (float*)carve(1024);
    const size_t actElems = (size_t)BN_ * SN_ * 128 * 128;
    _Float16* actA = (_Float16*)carve(actElems * sizeof(_Float16));
    _Float16* actB = (_Float16*)carve(actElems * sizeof(_Float16));

    // 1) FPS -> new_xyz directly into d_out
    const size_t fpsLds = (size_t)4 * NP_ * 4 + 256 * 4 + 256 * 4;
    fps_kernel<<<BN_, 256, fpsLds, stream>>>(xyz, dout);

    // 2) ball queries (read new_xyz from d_out)
    const size_t bqLds = (size_t)3 * NP_ * 4;
    ballq_kernel<16><<<BN_ * (SN_ / 8), 256, bqLds, stream>>>(xyz, dout, gidx0, 0.01f);
    ballq_kernel<32><<<BN_ * (SN_ / 8), 256, bqLds, stream>>>(xyz, dout, gidx1, 0.04f);
    ballq_kernel<128><<<BN_ * (SN_ / 8), 256, bqLds, stream>>>(xyz, dout, gidx2, 0.16f);

    // 3) MLP pipelines (WMMA GEMM + two-phase BN), fused BN+ReLU+max epilogue
    {
        const float* prm[12] = {P(0,0,0),P(0,0,1),P(0,0,2),P(0,0,3),
                                P(0,1,0),P(0,1,1),P(0,1,2),P(0,1,3),
                                P(0,2,0),P(0,2,1),P(0,2,2),P(0,2,3)};
        run_branch<16, 32, 32, 64, 0>(xyz, points, dout, gidx0, actA, actB,
                                      st0, st1, st2, prm, stream);
    }
    {
        const float* prm[12] = {P(1,0,0),P(1,0,1),P(1,0,2),P(1,0,3),
                                P(1,1,0),P(1,1,1),P(1,1,2),P(1,1,3),
                                P(1,2,0),P(1,2,1),P(1,2,2),P(1,2,3)};
        run_branch<32, 64, 64, 128, 64>(xyz, points, dout, gidx1, actA, actB,
                                        st0, st1, st2, prm, stream);
    }
    {
        const float* prm[12] = {P(2,0,0),P(2,0,1),P(2,0,2),P(2,0,3),
                                P(2,1,0),P(2,1,1),P(2,1,2),P(2,1,3),
                                P(2,2,0),P(2,2,1),P(2,2,2),P(2,2,3)};
        run_branch<128, 64, 96, 128, 192>(xyz, points, dout, gidx2, actA, actB,
                                          st0, st1, st2, prm, stream);
    }
}
